// SPECTRAGenerator_14508399526050
// MI455X (gfx1250) — matrix operand order repaired and checked by reading, not compiled, not run on hardware
//
#include <hip/hip_runtime.h>
#include <hip/hip_bf16.h>
#include <math.h>
#include <stdint.h>

typedef __bf16 bf16_t;
typedef __attribute__((ext_vector_type(16))) __bf16 v16bf;
typedef __attribute__((ext_vector_type(8)))  float  v8f;

typedef __attribute__((ext_vector_type(4))) unsigned int tdm_u32x4;
typedef __attribute__((ext_vector_type(8))) int          tdm_i32x8;
typedef __attribute__((ext_vector_type(4))) int          tdm_i32x4;

#define B_    32
#define T_    512
#define E_    300
#define EPAD  320
#define H_    256
#define G4H   1024
#define CMAX  155          // int(round(0.3*512)) + 1
#define TRR   10.0f        // TRANSITION / TEMP
#define INV_TEMP 100.0f
#define NEGV  (-1.0e9f)

union FragB16 { v16bf v; uint32_t u[8]; };

__device__ __forceinline__ float lse2(float a, float b) {
  float m = fmaxf(a, b);
  return m + __logf(__expf(a - m) + __expf(b - m));
}
__device__ __forceinline__ float sigm(float x) { return 1.0f / (1.0f + __expf(-x)); }

// ---------------------------------------------------------------------------
// Kernel 0a: transpose+convert weights to n-major bf16 (B-fragment friendly).
// ---------------------------------------------------------------------------
__global__ void prep_weights(const float* __restrict__ wi_f, const float* __restrict__ wi_b,
                             const float* __restrict__ wh_f, const float* __restrict__ wh_b,
                             bf16_t* __restrict__ wiT_f, bf16_t* __restrict__ wiT_b,
                             bf16_t* __restrict__ whT_f, bf16_t* __restrict__ whT_b) {
  int i = blockIdx.x * blockDim.x + threadIdx.x;
  if (i < G4H * EPAD) {
    int n = i / EPAD, k = i % EPAD;
    float vf = (k < E_) ? wi_f[(size_t)k * G4H + n] : 0.0f;
    float vb = (k < E_) ? wi_b[(size_t)k * G4H + n] : 0.0f;
    wiT_f[i] = (bf16_t)vf;
    wiT_b[i] = (bf16_t)vb;
  }
  if (i < G4H * H_) {
    int n = i / H_, k = i % H_;
    whT_f[i] = (bf16_t)wh_f[(size_t)k * G4H + n];
    whT_b[i] = (bf16_t)wh_b[(size_t)k * G4H + n];
  }
}

// ---------------------------------------------------------------------------
// Kernel 0b: gather embeddings -> bf16, K padded to 320 (zeros).
// ---------------------------------------------------------------------------
__global__ void gather_embed(const int* __restrict__ x, const float* __restrict__ embed,
                             bf16_t* __restrict__ emb_bf) {
  int i = blockIdx.x * blockDim.x + threadIdx.x;
  if (i >= B_ * T_ * EPAD) return;
  int row = i / EPAD, k = i % EPAD;
  float v = (k < E_) ? embed[(size_t)x[row] * E_ + k] : 0.0f;
  emb_bf[i] = (bf16_t)v;
}

// ---------------------------------------------------------------------------
// Kernel 1: xg = emb_bf @ WiT^T + b   (M=16384, K=320, N=1024)
// 8 waves share one 16-row A tile staged in LDS (TDM when available);
// each wave computes a 16Mx64N strip, 10 k-iters of v_wmma_f32_16x16x32_bf16.
// xg is written time-major: xg[t][b][4H] for scan-step locality.
// ---------------------------------------------------------------------------
__global__ void __launch_bounds__(256)
xg_gemm_kernel(const bf16_t* __restrict__ emb_bf, const bf16_t* __restrict__ wiT,
               const float* __restrict__ bias, float* __restrict__ xg) {
  __shared__ bf16_t atile[16 * EPAD];     // 10 KB
  const int lane = threadIdx.x & 31;
  const int wave = threadIdx.x >> 5;
  const int half = lane >> 4;
  const int l16  = lane & 15;
  const int mTile  = blockIdx.y;                 // 0..1023
  const int nGroup = blockIdx.x * 8 + wave;      // 0..15

#if __has_builtin(__builtin_amdgcn_tensor_load_to_lds) && __has_builtin(__builtin_amdgcn_s_wait_tensorcnt)
  // ---- Tensor Data Mover: one DMA of the contiguous 10 KB A tile ----
  if (wave == 0) {
    const uint64_t ga = (uint64_t)(uintptr_t)(emb_bf + (size_t)mTile * 16 * EPAD);
    const uint32_t lds = (uint32_t)(uintptr_t)(&atile[0]);   // LDS byte offset
    const unsigned NDW = (16 * EPAD * 2) / 4;                 // 2560 dwords
    tdm_u32x4 g0;
    g0.x = 1u;                                                // count=1 (valid)
    g0.y = lds;                                               // lds_addr [63:32]
    g0.z = (uint32_t)ga;                                      // global_addr lo
    g0.w = ((uint32_t)(ga >> 32) & 0x01FFFFFFu) | (2u << 30); // addr[56:32] | type=2
    tdm_i32x8 g1;
    g1[0] = (int)(2u << 16);        // data_size = 4B
    g1[1] = (int)(NDW << 16);       // tensor_dim0 lo16 (bits 63:48)
    g1[2] = (int)(1u << 16);        // tensor_dim0 hi=0; tensor_dim1 lo16 = 1
    g1[3] = (int)(NDW << 16);       // tensor_dim1 hi=0; tile_dim0 = NDW
    g1[4] = 1;                      // tile_dim1 = 1, tile_dim2 = 0
    g1[5] = (int)NDW;               // tensor_dim0_stride lo32
    g1[6] = 0;                      // stride hi, tensor_dim1_stride lo
    g1[7] = 0;
    tdm_i32x4 g2 = {0, 0, 0, 0};
    tdm_i32x4 g3 = {0, 0, 0, 0};
#if defined(__clang_major__) && (__clang_major__ >= 23)
    tdm_i32x8 g4 = {0, 0, 0, 0, 0, 0, 0, 0};
    __builtin_amdgcn_tensor_load_to_lds(g0, g1, g2, g3, g4, 0);
#else
    __builtin_amdgcn_tensor_load_to_lds(g0, g1, g2, g3, 0);
#endif
    __builtin_amdgcn_s_wait_tensorcnt((short)0);
  }
#else
  {  // fallback: cooperative dword copy (rows are contiguous)
    const uint32_t* src = (const uint32_t*)(emb_bf + (size_t)mTile * 16 * EPAD);
    uint32_t* dst = (uint32_t*)atile;
    for (int i = threadIdx.x; i < (16 * EPAD) / 2; i += 256) dst[i] = src[i];
  }
#endif
  __syncthreads();
  const uint32_t* at32 = (const uint32_t*)atile;

  v8f acc[4] = {};
#pragma unroll 2
  for (int kk = 0; kk < 10; ++kk) {
    const int kb = kk * 32;
    FragB16 a;
#pragma unroll
    for (int p = 0; p < 8; ++p) {
      // A map: dword p holds K = kb + 16*half + 2p, K+1
      int k = kb + 16 * half + 2 * p;
      a.u[p] = at32[(l16 * EPAD + k) >> 1];
    }
#pragma unroll
    for (int nt = 0; nt < 4; ++nt) {
      const int n = (nGroup * 4 + nt) * 16 + l16;
      const uint32_t* bsrc = (const uint32_t*)(wiT + (size_t)n * EPAD);
      FragB16 b;
#pragma unroll
      for (int p = 0; p < 8; ++p)           // B map: K = kb + 16*half + 2p, +1
        b.u[p] = bsrc[((kb + 16 * half) >> 1) + p];
      acc[nt] = __builtin_amdgcn_wmma_f32_16x16x32_bf16(
          false, a.v, false, b.v, (short)0, acc[nt], false, false);
    }
  }
#pragma unroll
  for (int nt = 0; nt < 4; ++nt) {
    const int n = (nGroup * 4 + nt) * 16 + l16;
    const float bv = bias[n];
#pragma unroll
    for (int r = 0; r < 8; ++r) {
      int m = mTile * 16 + r + half * 8;          // C map: M = r + 8*half
      int b_ = m >> 9;                            // m = b*T + t
      int t_ = m & (T_ - 1);
      xg[((size_t)t_ * B_ + b_) * G4H + n] = acc[nt][r] + bv;   // [t][b][4H]
    }
  }
}

// ---------------------------------------------------------------------------
// Kernel 2: LSTM scan. 2 blocks (dir), 1024 threads = 32 waves.
// Per step: G = h @ Wh via WMMA (M=32,K=256,N=1024). Wave w owns
// (mTile = w&1, hTile = w>>1) and its 4 gate tiles; cell state in registers.
// h kept in LDS as bf16. xg/hs are time-major: one contiguous slice per step.
// ---------------------------------------------------------------------------
__global__ void __launch_bounds__(1024)
lstm_scan_kernel(const float* __restrict__ xg_f, const float* __restrict__ xg_b,
                 const bf16_t* __restrict__ whT_f, const bf16_t* __restrict__ whT_b,
                 float* __restrict__ hs_f, float* __restrict__ hs_b) {
  const int dir = blockIdx.x;
  const float*  xg  = dir ? xg_b : xg_f;
  const bf16_t* whT = dir ? whT_b : whT_f;
  float*        hs  = dir ? hs_b : hs_f;

  __shared__ bf16_t hbuf[32 * 256];   // 16 KB, batch-major
  const uint32_t* hbuf32 = (const uint32_t*)hbuf;

  const int lane  = threadIdx.x & 31;
  const int wave  = threadIdx.x >> 5;
  const int half  = lane >> 4;
  const int l16   = lane & 15;
  const int mTile = wave & 1;
  const int hTile = wave >> 1;        // 0..15
  const int hcol  = hTile * 16 + l16;
  const int arow  = mTile * 16 + l16;

  for (int i = threadIdx.x; i < 32 * 256; i += 1024) hbuf[i] = (bf16_t)0.0f;
  float cst[8];
#pragma unroll
  for (int r = 0; r < 8; ++r) cst[r] = 0.0f;
  __syncthreads();

  for (int step = 0; step < T_; ++step) {
    const int t = dir ? (T_ - 1 - step) : step;
    if (step + 1 < T_) {   // L2 prefetch of next step's contiguous xg slice
      const int tn = dir ? (T_ - 2 - step) : (step + 1);
      const int brow0 = mTile * 16 + half * 8;
      __builtin_prefetch(&xg[((size_t)tn * B_ + brow0) * G4H + hcol], 0, 1);
    }
    v8f acc[4] = {};
#pragma unroll 2
    for (int kk = 0; kk < 8; ++kk) {
      const int kb = kk * 32;
      FragB16 a;
#pragma unroll
      for (int p = 0; p < 8; ++p) {
        int k = kb + 16 * half + 2 * p;
        a.u[p] = hbuf32[(arow * 256 + k) >> 1];
      }
#pragma unroll
      for (int g = 0; g < 4; ++g) {
        const int n = g * H_ + hcol;
        const uint32_t* bsrc = (const uint32_t*)(whT + (size_t)n * H_);
        FragB16 b;
#pragma unroll
        for (int p = 0; p < 8; ++p)
          b.u[p] = bsrc[((kb + 16 * half) >> 1) + p];
        acc[g] = __builtin_amdgcn_wmma_f32_16x16x32_bf16(
            false, a.v, false, b.v, (short)0, acc[g], false, false);
      }
    }
    // gates + state update, in-register on C fragments
    float hnew[8];
#pragma unroll
    for (int r = 0; r < 8; ++r) {
      const int brow = mTile * 16 + r + half * 8;           // batch index
      const size_t base = ((size_t)t * B_ + brow) * G4H;    // [t][b][4H]
      float gi = acc[0][r] + xg[base + 0 * H_ + hcol];
      float gf = acc[1][r] + xg[base + 1 * H_ + hcol];
      float gg = acc[2][r] + xg[base + 2 * H_ + hcol];
      float go = acc[3][r] + xg[base + 3 * H_ + hcol];
      float cv = sigm(gf) * cst[r] + sigm(gi) * tanhf(gg);
      cst[r] = cv;
      hnew[r] = sigm(go) * tanhf(cv);
    }
    __syncthreads();   // all A-fragment reads of hbuf done
#pragma unroll
    for (int r = 0; r < 8; ++r) {
      const int brow = mTile * 16 + r + half * 8;
      hbuf[brow * 256 + hcol] = (bf16_t)hnew[r];
      hs[((size_t)t * B_ + brow) * H_ + hcol] = hnew[r];    // [t][b][H]
    }
    __syncthreads();   // h published for next step
  }
}

// ---------------------------------------------------------------------------
// Kernel 3: u0[b,t] = (concat(h_f,h_b) . w_out + b_out) / TEMP
// hs layout is [t][b][H].
// ---------------------------------------------------------------------------
__global__ void score_kernel(const float* __restrict__ hs_f, const float* __restrict__ hs_b,
                             const float* __restrict__ w_out, const float* __restrict__ b_out,
                             float* __restrict__ u0) {
  int r = blockIdx.x * blockDim.x + threadIdx.x;
  if (r >= B_ * T_) return;
  int b = r / T_, t = r % T_;
  const float4* hf = (const float4*)(hs_f + ((size_t)t * B_ + b) * H_);
  const float4* hb = (const float4*)(hs_b + ((size_t)t * B_ + b) * H_);
  const float4* wf = (const float4*)(w_out);
  const float4* wb = (const float4*)(w_out + H_);
  float s = b_out[0];
  for (int i = 0; i < H_ / 4; ++i) {
    float4 a = hf[i], w = wf[i];
    s = fmaf(a.x, w.x, fmaf(a.y, w.y, fmaf(a.z, w.z, fmaf(a.w, w.w, s))));
    float4 c = hb[i], v = wb[i];
    s = fmaf(c.x, v.x, fmaf(c.y, v.y, fmaf(c.z, v.z, fmaf(c.w, v.w, s))));
  }
  u0[r] = s * INV_TEMP;
}

// ---------------------------------------------------------------------------
// Kernel 4: CRF forward, storing alpha[t] for the adjoint sweep.
// ---------------------------------------------------------------------------
__global__ void __launch_bounds__(256)
crf_forward_kernel(const float* __restrict__ u0, float* __restrict__ astore) {
  const int b = blockIdx.x;
  const int c = threadIdx.x;
  __shared__ float a0[CMAX], a1[CMAX];
  const float* u = u0 + (size_t)b * T_;
  float* st = astore + (size_t)b * T_ * (2 * CMAX);
  if (c < CMAX) {
    a0[c] = (c == 1) ? (u[0] + TRR) : NEGV;
    a1[c] = (c == 0) ? 0.0f : NEGV;
    st[c * 2 + 0] = a0[c];
    st[c * 2 + 1] = a1[c];
  }
  __syncthreads();
  for (int s = 0; s < T_ - 1; ++s) {
    float n0 = NEGV, n1 = NEGV;
    if (c < CMAX) {
      float a0p = (c >= 1) ? a0[c - 1] : NEGV;
      float a1p = (c >= 1) ? a1[c - 1] : NEGV;
      n0 = u[s + 1] + lse2(a0p + TRR, a1p);
      n1 = lse2(a0[c], a1[c]);
    }
    __syncthreads();
    if (c < CMAX) {
      a0[c] = n0; a1[c] = n1;
      float* slot = st + (size_t)(s + 1) * (2 * CMAX);
      slot[c * 2 + 0] = n0;
      slot[c * 2 + 1] = n1;
    }
    __syncthreads();
  }
}

// ---------------------------------------------------------------------------
// Kernel 5: CRF adjoint sweep -> out[b,t] = d logZ / d u0[b,t]
// ---------------------------------------------------------------------------
__global__ void __launch_bounds__(256)
crf_backward_kernel(const float* __restrict__ u0, const float* __restrict__ astore,
                    float* __restrict__ out) {
  const int b = blockIdx.x;
  const int c = threadIdx.x;
  __shared__ float b0[CMAX + 1], b1[CMAX];
  __shared__ float red[256];
  const float* u  = u0 + (size_t)b * T_;
  const float* st = astore + (size_t)b * T_ * (2 * CMAX);
  float* z = out + (size_t)b * T_;

  // softmax over final states (budget=154 => all c allowed)
  const float* fin = st + (size_t)(T_ - 1) * (2 * CMAX);
  float f0 = NEGV, f1 = NEGV;
  if (c < CMAX) { f0 = fin[c * 2 + 0] + TRR; f1 = fin[c * 2 + 1]; }
  red[c] = fmaxf(f0, f1);
  __syncthreads();
  for (int off = 128; off > 0; off >>= 1) {
    if (c < off) red[c] = fmaxf(red[c], red[c + off]);
    __syncthreads();
  }
  float mx = red[0];
  __syncthreads();
  red[c] = (c < CMAX) ? (__expf(f0 - mx) + __expf(f1 - mx)) : 0.0f;
  __syncthreads();
  for (int off = 128; off > 0; off >>= 1) {
    if (c < off) red[c] += red[c + off];
    __syncthreads();
  }
  float lz = mx + __logf(red[0]);
  __syncthreads();
  if (c < CMAX) { b0[c] = __expf(f0 - lz); b1[c] = __expf(f1 - lz); }
  if (c == 0) b0[CMAX] = 0.0f;     // pad for c+1 access
  __syncthreads();

  for (int s = T_ - 2; s >= 0; --s) {
    const float* slot = st + (size_t)s * (2 * CMAX);
    float ai0 = NEGV, ai1 = NEGV;
    if (c < CMAX) { ai0 = slot[c * 2 + 0]; ai1 = slot[c * 2 + 1]; }
    // grad u0[s+1] = sum_c bar(n0[c])
    red[c] = (c < CMAX) ? b0[c] : 0.0f;
    __syncthreads();
    for (int off = 128; off > 0; off >>= 1) {
      if (c < off) red[c] += red[c + off];
      __syncthreads();
    }
    if (c == 0) z[s + 1] = red[0];
    // adjoint of step inputs
    float t0 = 0.0f, t1 = 0.0f;
    if (c < CMAX) {
      float n1v = lse2(ai0, ai1);
      t0 = b1[c] * __expf(ai0 - n1v);
      t1 = b1[c] * __expf(ai1 - n1v);
      float m  = lse2(ai0 + TRR, ai1);   // = n0[c+1] - u
      float bn = b0[c + 1];
      t0 += bn * __expf(ai0 + TRR - m);
      t1 += bn * __expf(ai1 - m);
    }
    __syncthreads();
    if (c < CMAX) { b0[c] = t0; b1[c] = t1; }
    __syncthreads();
  }
  if (c == 0) z[0] = b0[1];   // adjoint of init: alpha[1,0] = u0[0] + TR
}

// ---------------------------------------------------------------------------
extern "C" void kernel_launch(void* const* d_in, const int* in_sizes, int n_in,
                              void* d_out, int out_size, void* d_ws, size_t ws_size,
                              hipStream_t stream) {
  (void)in_sizes; (void)n_in; (void)out_size; (void)ws_size;
  const int*   x     = (const int*)d_in[0];
  const float* embed = (const float*)d_in[3];
  const float* wi_f  = (const float*)d_in[4];
  const float* wh_f  = (const float*)d_in[5];
  const float* bf    = (const float*)d_in[6];
  const float* wi_b  = (const float*)d_in[7];
  const float* wh_b  = (const float*)d_in[8];
  const float* bb    = (const float*)d_in[9];
  const float* w_out = (const float*)d_in[10];
  const float* b_out = (const float*)d_in[11];
  float* out = (float*)d_out;

  char* ws = (char*)d_ws;
  size_t off = 0;
  auto alloc = [&](size_t bytes) -> void* {
    void* p = ws + off;
    off = (off + bytes + 255) & ~(size_t)255;
    return p;
  };
  float*  xg_f    = (float*)alloc((size_t)B_ * T_ * G4H * sizeof(float));   // 64 MB
  float*  xg_b    = (float*)alloc((size_t)B_ * T_ * G4H * sizeof(float));   // 64 MB
  float*  hs_f    = (float*)alloc((size_t)B_ * T_ * H_ * sizeof(float));    // 16 MB
  float*  hs_b    = (float*)alloc((size_t)B_ * T_ * H_ * sizeof(float));    // 16 MB
  float*  u0      = (float*)alloc((size_t)B_ * T_ * sizeof(float));
  bf16_t* emb_bf  = (bf16_t*)alloc((size_t)B_ * T_ * EPAD * sizeof(bf16_t)); // 10.5 MB
  bf16_t* wiT_f   = (bf16_t*)alloc((size_t)G4H * EPAD * sizeof(bf16_t));
  bf16_t* wiT_b   = (bf16_t*)alloc((size_t)G4H * EPAD * sizeof(bf16_t));
  bf16_t* whT_f   = (bf16_t*)alloc((size_t)G4H * H_ * sizeof(bf16_t));
  bf16_t* whT_b   = (bf16_t*)alloc((size_t)G4H * H_ * sizeof(bf16_t));
  float*  astore  = (float*)alloc((size_t)B_ * T_ * 2 * CMAX * sizeof(float)); // ~20 MB

  // 0: weight prep + embedding gather (bf16, padded, fragment-friendly)
  prep_weights<<<(G4H * EPAD + 255) / 256, 256, 0, stream>>>(
      wi_f, wi_b, wh_f, wh_b, wiT_f, wiT_b, whT_f, whT_b);
  gather_embed<<<(B_ * T_ * EPAD + 255) / 256, 256, 0, stream>>>(x, embed, emb_bf);

  // 1: input projections (WMMA GEMM, TDM-staged A tiles), both directions
  dim3 ggrid(2, (B_ * T_) / 16, 1);
  xg_gemm_kernel<<<ggrid, 256, 0, stream>>>(emb_bf, wiT_f, bf, xg_f);
  xg_gemm_kernel<<<ggrid, 256, 0, stream>>>(emb_bf, wiT_b, bb, xg_b);

  // 2: bidirectional LSTM scan (WMMA recurrence)
  lstm_scan_kernel<<<2, 1024, 0, stream>>>(xg_f, xg_b, whT_f, whT_b, hs_f, hs_b);

  // 3: scores -> u0
  score_kernel<<<(B_ * T_ + 255) / 256, 256, 0, stream>>>(hs_f, hs_b, w_out, b_out, u0);

  // 4/5: constrained-CRF forward + adjoint -> marginals (the output)
  crf_forward_kernel<<<B_, 256, 0, stream>>>(u0, astore);
  crf_backward_kernel<<<B_, 256, 0, stream>>>(u0, astore, out);
}
